// MambaEncoderLayer_43465069036251
// MI455X (gfx1250) — matrix operand order repaired and checked by reading, not compiled, not run on hardware
//
#include <hip/hip_runtime.h>
#include <hip/hip_bf16.h>

// ---------------- problem constants (from reference) ----------------
#define EMBED   128
#define DINNER  256
#define DSTATE  16
#define DTRANK  8
#define KDIR    4
#define BSZ     4
#define Hc      64
#define Wc      64
#define Lc      4096         // Hc*Wc
#define FFC     512

// ---------------- WMMA bf16 helpers (gfx1250, wave32) ----------------
typedef __bf16 v16bf __attribute__((ext_vector_type(16)));
typedef float  v8f   __attribute__((ext_vector_type(8)));

__device__ __forceinline__ v8f wmma_bf16(v16bf a, v16bf b, v8f c) {
  // D = A(16x32) * B(32x16) + C, f32 accumulate
  return __builtin_amdgcn_wmma_f32_16x16x32_bf16(false, a, false, b, (short)0, c,
                                                 false, false);
}

// A tile 16x32 (MxK) from bf16 memory, row-major, leading dim lda.
// Lane L(0..15): row M=L, K 0-15 ; lane L+16: row M=L, K 16-31.
__device__ __forceinline__ v16bf load_A16(const __bf16* __restrict__ A, int lda) {
  const int lane = threadIdx.x & 31;
  const int m = lane & 15;
  const int kh = (lane >> 4) << 4;       // 0 or 16
  const __bf16* row = A + (size_t)m * lda + kh;
  v16bf r;
#pragma unroll
  for (int e = 0; e < 8; ++e) r[e] = row[e];
#pragma unroll
  for (int e = 0; e < 8; ++e) r[8 + e] = row[8 + e];
  return r;
}

// B tile 32x16 (KxN), memory row-major (N,K) with leading dim ldb (token-major).
__device__ __forceinline__ v16bf load_Bnk16(const __bf16* __restrict__ Bp, int ldb) {
  const int lane = threadIdx.x & 31;
  const int n = lane & 15;
  const int kh = (lane >> 4) << 4;       // 0 or 16
  const __bf16* row = Bp + (size_t)n * ldb + kh;
  v16bf r;
#pragma unroll
  for (int e = 0; e < 16; ++e) r[e] = row[e];
  return r;
}

// B tile 32x16 (KxN), memory row-major (K,N) with leading dim ldb.
__device__ __forceinline__ v16bf load_Bkn16(const __bf16* __restrict__ Bp, int ldb) {
  const int lane = threadIdx.x & 31;
  const int n = lane & 15;
  const int kh = (lane >> 4) << 4;
  v16bf r;
#pragma unroll
  for (int e = 0; e < 16; ++e) r[e] = Bp[(size_t)(kh + e) * ldb + n];
  return r;
}

// ---------------- prep: f32 -> bf16 converters ----------------
__global__ void k_cvt_bf16(const float* __restrict__ src, __bf16* __restrict__ dst,
                           int n) {
  const int i = blockIdx.x * 256 + threadIdx.x;
  if (i < n) dst[i] = (__bf16)src[i];
}

// x_proj_w (K,40,256) -> padded bf16 (K,48,256), rows 40..47 zero
__global__ void k_cvt_xpw(const float* __restrict__ src, __bf16* __restrict__ dst) {
  const int i = blockIdx.x * 256 + threadIdx.x;     // K*48*256 = 49152
  const int c = i & 255;
  int t = i >> 8;
  const int m = t % 48;
  const int k = t / 48;
  dst[i] = (m < 40) ? (__bf16)src[((size_t)k * 40 + m) * DINNER + c] : (__bf16)0.f;
}

// ---------------- GEMM, out channel-major f32 ----------------
// Out[b][m][n] = sum_k A[m][k] * B[b][n][k]  (+bias[m])
// grid: (Lc/64, M/64, BSZ), block 128 (4 waves, each 32x32).
__global__ void k_gemm_ml(const __bf16* __restrict__ A, const __bf16* __restrict__ Bb,
                          const float* __restrict__ bias, float* __restrict__ Out,
                          int Kd, size_t bStride, size_t cStride) {
  const int b = blockIdx.z;
  const int Mtile = blockIdx.y * 64, Ntile = blockIdx.x * 64;
  const int warp = threadIdx.x >> 5;
  const int wm = (warp >> 1) * 32, wn = (warp & 1) * 32;
  const __bf16* Bp = Bb + (size_t)b * bStride;
  v8f acc[2][2] = {};
  for (int kk = 0; kk < Kd; kk += 32) {
    v16bf a0 = load_A16(A + (size_t)(Mtile + wm) * Kd + kk, Kd);
    v16bf a1 = load_A16(A + (size_t)(Mtile + wm + 16) * Kd + kk, Kd);
    v16bf b0 = load_Bnk16(Bp + (size_t)(Ntile + wn) * Kd + kk, Kd);
    v16bf b1 = load_Bnk16(Bp + (size_t)(Ntile + wn + 16) * Kd + kk, Kd);
    acc[0][0] = wmma_bf16(a0, b0, acc[0][0]);
    acc[0][1] = wmma_bf16(a0, b1, acc[0][1]);
    acc[1][0] = wmma_bf16(a1, b0, acc[1][0]);
    acc[1][1] = wmma_bf16(a1, b1, acc[1][1]);
  }
  const int lane = threadIdx.x & 31;
  const int n = lane & 15, mh = (lane >> 4) * 8;
  float* ob = Out + (size_t)b * cStride;
#pragma unroll
  for (int i = 0; i < 2; ++i)
#pragma unroll
    for (int j = 0; j < 2; ++j) {
      const int m0 = Mtile + wm + i * 16 + mh;
      const int n0 = Ntile + wn + j * 16 + n;
#pragma unroll
      for (int r = 0; r < 8; ++r) {
        float v = acc[i][j][r];
        if (bias) v += bias[m0 + r];
        ob[(size_t)(m0 + r) * Lc + n0] = v;
      }
    }
}

// ---------------- depthwise 3x3 conv + activation ----------------
// act: 0 = SiLU, 1 = exact GELU.  Optional f32 and/or bf16 outputs.
__global__ void k_dwconv_act(const float* __restrict__ In, const float* __restrict__ W,
                             const float* __restrict__ Bias, float* __restrict__ OutF,
                             __bf16* __restrict__ OutH, int C, size_t inBStride,
                             size_t outBStride, int act) {
  const int idx = blockIdx.x * blockDim.x + threadIdx.x;
  const int wcol = idx & 63;
  int t = idx >> 6;
  const int hrow = t & 63; t >>= 6;
  const int c = t % C;
  const int b = t / C;
  const float* ip = In + (size_t)b * inBStride + (size_t)c * (Hc * Wc);
  const float* wp = W + c * 9;
  float s = Bias[c];
#pragma unroll
  for (int dy = -1; dy <= 1; ++dy) {
    const int hh = hrow + dy;
    if ((unsigned)hh >= (unsigned)Hc) continue;
#pragma unroll
    for (int dx = -1; dx <= 1; ++dx) {
      const int ww = wcol + dx;
      if ((unsigned)ww >= (unsigned)Wc) continue;
      s = fmaf(ip[hh * Wc + ww], wp[(dy + 1) * 3 + (dx + 1)], s);
    }
  }
  if (act == 0)       s = s / (1.f + __expf(-s));                       // SiLU
  else                s = 0.5f * s * (1.f + erff(s * 0.70710678118f));  // GELU
  const size_t o = (size_t)b * outBStride + (size_t)c * (Hc * Wc) + hrow * Wc + wcol;
  if (OutF) OutF[o] = s;
  if (OutH) OutH[o] = (__bf16)s;
}

// ---------------- cross-scan index map (k=0 row, 1 col, 2 rev-row, 3 rev-col) --
__device__ __forceinline__ int scan_map(int k, int l) {
  if (k >= 2) l = Lc - 1 - l;
  if (k & 1) l = (l & (Hc - 1)) * Wc + (l >> 6);
  return l;
}

// ---------------- x_proj GEMM (M=40 padded to 48) ----------------
// x_dbl[b,k,c,l] = sum_d x_proj_w[k,c,d] * xm[b,d,map_k(l)]
// grid: (Lc/64, BSZ*KDIR), block 128 (4 waves, each 48x16)
__global__ void k_gemm_xproj(const __bf16* __restrict__ Xpw,
                             const __bf16* __restrict__ Xm16,
                             float* __restrict__ Xdbl) {
  const int bk = blockIdx.y;
  const int b = bk >> 2, k = bk & 3;
  const int warp = threadIdx.x >> 5;
  const int lane = threadIdx.x & 31;
  const int Ng = blockIdx.x * 64 + warp * 16;
  const __bf16* A = Xpw + (size_t)k * 48 * DINNER;
  const int lm = scan_map(k, Ng + (lane & 15));
  const __bf16* xcol = Xm16 + (size_t)b * DINNER * Lc + lm;
  const int kh = (lane >> 4) << 4;
  v8f acc[3] = {};
  for (int kk = 0; kk < DINNER; kk += 32) {
    v16bf bt;
#pragma unroll
    for (int e = 0; e < 16; ++e) bt[e] = xcol[(size_t)(kk + kh + e) * Lc];
    v16bf a0 = load_A16(A + kk, DINNER);
    v16bf a1 = load_A16(A + (size_t)16 * DINNER + kk, DINNER);
    v16bf a2 = load_A16(A + (size_t)32 * DINNER + kk, DINNER);
    acc[0] = wmma_bf16(a0, bt, acc[0]);
    acc[1] = wmma_bf16(a1, bt, acc[1]);
    acc[2] = wmma_bf16(a2, bt, acc[2]);
  }
  const int n = lane & 15, mh = (lane >> 4) * 8;
  float* ob = Xdbl + (size_t)bk * 40 * Lc + Ng + n;
#pragma unroll
  for (int i = 0; i < 3; ++i)
#pragma unroll
    for (int r = 0; r < 8; ++r) {
      const int m = i * 16 + mh + r;
      if (m < 40) ob[(size_t)m * Lc] = acc[i][r];
    }
}

// ---------------- dt_proj (K=8) + softplus ----------------
__global__ void k_dtproj(const float* __restrict__ Xdbl, const float* __restrict__ Wdt,
                         const float* __restrict__ Bdt, float* __restrict__ Delta) {
  const int idx = blockIdx.x * 256 + threadIdx.x;
  const int l = idx & (Lc - 1);
  int t = idx >> 12;
  const int d = t & (DINNER - 1); t >>= 8;
  const int k = t & 3;
  const int b = t >> 2;
  const int bk = b * KDIR + k;
  const float* dts = Xdbl + (size_t)bk * 40 * Lc + l;        // rows 0..7
  const float* wp  = Wdt + (size_t)(k * DINNER + d) * DTRANK;
  float s = Bdt[k * DINNER + d];
#pragma unroll
  for (int r = 0; r < DTRANK; ++r) s = fmaf(wp[r], dts[(size_t)r * Lc], s);
  s = (s > 20.f) ? s : log1pf(__expf(s));                    // softplus
  Delta[((size_t)bk * DINNER + d) * Lc + l] = s;
}

// ---------------- selective scan + cross-merge ----------------
// One block per (b,k); one thread per channel d; h[16] in registers.
// Shared B/C vectors staged via async global->LDS copies (ASYNCcnt fenced).
__global__ void k_scan(const float* __restrict__ Xm, const float* __restrict__ Xdbl,
                       const float* __restrict__ Delta, const float* __restrict__ Alogs,
                       const float* __restrict__ Ds, float* __restrict__ Ym) {
  const int bk = blockIdx.x;
  const int b = bk >> 2, k = bk & 3;
  const int d = threadIdx.x;                 // 0..255
  const float* xp  = Xm + ((size_t)b * DINNER + d) * Lc;
  const float* dp  = Delta + ((size_t)bk * DINNER + d) * Lc;
  const float* xdb = Xdbl + (size_t)bk * 40 * Lc;
  float* yb = Ym + (size_t)b * Lc * DINNER + d;
  const int kd = k * DINNER + d;
  float An[DSTATE], h[DSTATE];
#pragma unroll
  for (int n = 0; n < DSTATE; ++n) { An[n] = -__expf(Alogs[kd * DSTATE + n]); h[n] = 0.f; }
  const float dskip = Ds[kd];
  __shared__ float sBC[64 * 32];             // [step][n: 0..15 = B, 16..31 = C]
  for (int l0 = 0; l0 < Lc; l0 += 64) {
    __syncthreads();
    // async copy x_dbl rows 8..39 (B then C), 64 steps, into LDS (transposed)
#pragma unroll
    for (int it = 0; it < 8; ++it) {
      const int i = threadIdx.x + it * 256;  // 0..2047, lanes walk s (coalesced)
      const int r = i >> 6, s = i & 63;
      const float* gp = xdb + (size_t)(8 + r) * Lc + l0 + s;
      const unsigned ldsoff = (unsigned)(size_t)&sBC[s * 32 + r];
      asm volatile("global_load_async_to_lds_b32 %0, %1, off"
                   :: "v"(ldsoff), "v"(gp) : "memory");
    }
    if (l0 + 128 < Lc) __builtin_prefetch(dp + l0 + 128, 0, 0);
    asm volatile("s_wait_asynccnt 0x0" ::: "memory");
    __syncthreads();
    for (int s = 0; s < 64; ++s) {
      const int l = l0 + s;
      const float dt = dp[l];
      const int lm = scan_map(k, l);
      const float u = xp[lm];
      const float du = dt * u;
      const float* bc = &sBC[s * 32];
      float acc = 0.f;
#pragma unroll
      for (int n = 0; n < DSTATE; ++n) {
        h[n] = __expf(dt * An[n]) * h[n] + du * bc[n];
        acc = fmaf(h[n], bc[16 + n], acc);
      }
      atomicAdd(yb + (size_t)lm * DINNER, acc + dskip * u);  // cross-merge
    }
  }
}

// ---------------- out_norm + SiLU(z) gating -> bf16 ----------------
__global__ void k_ln_gate(const float* __restrict__ Ym, const float* __restrict__ Xz,
                          const float* __restrict__ g, const float* __restrict__ bb,
                          __bf16* __restrict__ Out) {
  const int warp = threadIdx.x >> 5, lane = threadIdx.x & 31;
  const int bl = blockIdx.x * 8 + warp;
  const int b = bl / Lc, l = bl % Lc;
  const float* row = Ym + (size_t)bl * DINNER;
  float v[8], s = 0.f, sq = 0.f;
#pragma unroll
  for (int i = 0; i < 8; ++i) { v[i] = row[lane + 32 * i]; s += v[i]; sq += v[i] * v[i]; }
#pragma unroll
  for (int off = 16; off; off >>= 1) { s += __shfl_xor(s, off); sq += __shfl_xor(sq, off); }
  const float mean = s * (1.f / DINNER);
  const float rstd = rsqrtf(sq * (1.f / DINNER) - mean * mean + 1e-5f);
  const float* zp = Xz + ((size_t)b * (2 * DINNER) + DINNER) * Lc + l;
#pragma unroll
  for (int i = 0; i < 8; ++i) {
    const int dd = lane + 32 * i;
    const float t = (v[i] - mean) * rstd * g[dd] + bb[dd];
    const float zv = zp[(size_t)dd * Lc];
    Out[(size_t)bl * DINNER + dd] = (__bf16)(t * (zv / (1.f + __expf(-zv))));
  }
}

// ---------------- out_proj GEMM + residual (token-major store) ----------------
// xres[b,l,c] = x[b,l,c] + sum_d W[c,d] * yg[b,l,d]
__global__ void k_gemm_outproj(const __bf16* __restrict__ Wp,
                               const __bf16* __restrict__ Yg,
                               const float* __restrict__ Xin, float* __restrict__ Out) {
  const int b = blockIdx.z;
  const int Mtile = blockIdx.y * 64, Ntile = blockIdx.x * 64;
  const int warp = threadIdx.x >> 5;
  const int wm = (warp >> 1) * 32, wn = (warp & 1) * 32;
  const __bf16* Bp = Yg + (size_t)b * Lc * DINNER;
  v8f acc[2][2] = {};
  for (int kk = 0; kk < DINNER; kk += 32) {
    v16bf a0 = load_A16(Wp + (size_t)(Mtile + wm) * DINNER + kk, DINNER);
    v16bf a1 = load_A16(Wp + (size_t)(Mtile + wm + 16) * DINNER + kk, DINNER);
    v16bf b0 = load_Bnk16(Bp + (size_t)(Ntile + wn) * DINNER + kk, DINNER);
    v16bf b1 = load_Bnk16(Bp + (size_t)(Ntile + wn + 16) * DINNER + kk, DINNER);
    acc[0][0] = wmma_bf16(a0, b0, acc[0][0]);
    acc[0][1] = wmma_bf16(a0, b1, acc[0][1]);
    acc[1][0] = wmma_bf16(a1, b0, acc[1][0]);
    acc[1][1] = wmma_bf16(a1, b1, acc[1][1]);
  }
  const int lane = threadIdx.x & 31;
  const int n = lane & 15, mh = (lane >> 4) * 8;
#pragma unroll
  for (int i = 0; i < 2; ++i)
#pragma unroll
    for (int j = 0; j < 2; ++j)
#pragma unroll
      for (int r = 0; r < 8; ++r) {
        const int m = Mtile + wm + i * 16 + mh + r;
        const int l = Ntile + wn + j * 16 + n;
        const size_t o = ((size_t)b * Lc + l) * EMBED + m;
        Out[o] = Xin[o] + acc[i][j][r];
      }
}

// ---------------- norm2 (128 ch, wave per token) -> bf16 ----------------
__global__ void k_ln2(const float* __restrict__ X, const float* __restrict__ g,
                      const float* __restrict__ bb, __bf16* __restrict__ Out) {
  const int warp = threadIdx.x >> 5, lane = threadIdx.x & 31;
  const int bl = blockIdx.x * 8 + warp;
  const float* row = X + (size_t)bl * EMBED;
  float v[4], s = 0.f, sq = 0.f;
#pragma unroll
  for (int i = 0; i < 4; ++i) { v[i] = row[lane + 32 * i]; s += v[i]; sq += v[i] * v[i]; }
#pragma unroll
  for (int off = 16; off; off >>= 1) { s += __shfl_xor(s, off); sq += __shfl_xor(sq, off); }
  const float mean = s * (1.f / EMBED);
  const float rstd = rsqrtf(sq * (1.f / EMBED) - mean * mean + 1e-5f);
#pragma unroll
  for (int i = 0; i < 4; ++i) {
    const int dd = lane + 32 * i;
    Out[(size_t)bl * EMBED + dd] = (__bf16)((v[i] - mean) * rstd * g[dd] + bb[dd]);
  }
}

// ---------------- fc2 GEMM + bias + residual -> d_out ----------------
// out[b,l,c] = xres[b,l,c] + fc2_b[c] + sum_f W[c,f] * g[b,f,l]
__global__ void k_gemm_fc2(const __bf16* __restrict__ Wp, const __bf16* __restrict__ G,
                           const float* __restrict__ Xres, const float* __restrict__ bias,
                           float* __restrict__ Out) {
  const int b = blockIdx.z;
  const int Mtile = blockIdx.y * 64, Ntile = blockIdx.x * 64;
  const int warp = threadIdx.x >> 5;
  const int wm = (warp >> 1) * 32, wn = (warp & 1) * 32;
  const __bf16* Bp = G + (size_t)b * FFC * Lc;
  v8f acc[2][2] = {};
  for (int kk = 0; kk < FFC; kk += 32) {
    v16bf a0 = load_A16(Wp + (size_t)(Mtile + wm) * FFC + kk, FFC);
    v16bf a1 = load_A16(Wp + (size_t)(Mtile + wm + 16) * FFC + kk, FFC);
    v16bf b0 = load_Bkn16(Bp + (size_t)kk * Lc + Ntile + wn, Lc);
    v16bf b1 = load_Bkn16(Bp + (size_t)kk * Lc + Ntile + wn + 16, Lc);
    acc[0][0] = wmma_bf16(a0, b0, acc[0][0]);
    acc[0][1] = wmma_bf16(a0, b1, acc[0][1]);
    acc[1][0] = wmma_bf16(a1, b0, acc[1][0]);
    acc[1][1] = wmma_bf16(a1, b1, acc[1][1]);
  }
  const int lane = threadIdx.x & 31;
  const int n = lane & 15, mh = (lane >> 4) * 8;
#pragma unroll
  for (int i = 0; i < 2; ++i)
#pragma unroll
    for (int j = 0; j < 2; ++j)
#pragma unroll
      for (int r = 0; r < 8; ++r) {
        const int m = Mtile + wm + i * 16 + mh + r;
        const int l = Ntile + wn + j * 16 + n;
        const size_t o = ((size_t)b * Lc + l) * EMBED + m;
        Out[o] = Xres[o] + bias[m] + acc[i][j][r];
      }
}

// ---------------- host-side orchestration ----------------
extern "C" void kernel_launch(void* const* d_in, const int* in_sizes, int n_in,
                              void* d_out, int out_size, void* d_ws, size_t ws_size,
                              hipStream_t stream) {
  const float* x          = (const float*)d_in[0];
  const float* in_proj_w  = (const float*)d_in[1];
  const float* conv_w     = (const float*)d_in[2];
  const float* conv_b     = (const float*)d_in[3];
  const float* x_proj_w   = (const float*)d_in[4];
  const float* dt_proj_w  = (const float*)d_in[5];
  const float* dt_proj_b  = (const float*)d_in[6];
  const float* A_logs     = (const float*)d_in[7];
  const float* Ds         = (const float*)d_in[8];
  const float* out_norm_g = (const float*)d_in[9];
  const float* out_norm_b = (const float*)d_in[10];
  const float* out_proj_w = (const float*)d_in[11];
  const float* norm2_g    = (const float*)d_in[12];
  const float* norm2_b    = (const float*)d_in[13];
  const float* fc1_w      = (const float*)d_in[14];
  const float* fc1_b      = (const float*)d_in[15];
  const float* pe_w       = (const float*)d_in[16];
  const float* pe_b       = (const float*)d_in[17];
  const float* fc2_w      = (const float*)d_in[18];
  const float* fc2_b      = (const float*)d_in[19];
  float* outp = (float*)d_out;

  // workspace layout (byte offsets)
  char* base = (char*)d_ws;
  float*  xz    = (float*)(base);                    // (B,512,L) f32  33,554,432 B
  float*  xm    = (float*)(base + 33554432);         // (B,256,L) f32  16,777,216 B
  float*  xdbl  = (float*)(base + 50331648);         // (B*K,40,L) f32 10,485,760 B
  float*  delta = (float*)(base + 60817408);         // (B*K,256,L)    67,108,864 B
  float*  ym    = (float*)(base + 127926272);        // (B,L,256) f32  16,777,216 B
  float*  xres  = (float*)(base + 144703488);        // (B,L,128) f32   8,388,608 B
  __bf16* yg16  = (__bf16*)(base + 153092096);       // (B,L,256) bf16  8,388,608 B
  __bf16* xn16  = (__bf16*)(base + 161480704);       // (B,L,128) bf16  4,194,304 B
  __bf16* xb16  = (__bf16*)(base + 165675008);       // (B,L,128) bf16  4,194,304 B
  __bf16* xm16  = (__bf16*)(base + 169869312);       // (B,256,L) bf16  8,388,608 B
  __bf16* wib   = (__bf16*)(base + 178257920);       // in_proj_w bf16
  __bf16* wxp   = (__bf16*)(base + 178388992);       // x_proj_w padded (K,48,256)
  __bf16* wop   = (__bf16*)(base + 178487296);       // out_proj_w bf16
  __bf16* wf1   = (__bf16*)(base + 178552832);       // fc1_w bf16
  __bf16* wf2   = (__bf16*)(base + 178683904);       // fc2_w bf16
  float*  ffn1  = delta;                             // reuse: (B,512,L) f32
  __bf16* ffn2  = (__bf16*)((char*)delta + 33554432);// reuse: (B,512,L) bf16

  // ---- prep: bf16 conversions (weights + input activations) ----
  k_cvt_bf16<<<(BSZ * Lc * EMBED) / 256, 256, 0, stream>>>(x, xb16, BSZ * Lc * EMBED);
  k_cvt_bf16<<<256, 256, 0, stream>>>(in_proj_w, wib, 2 * DINNER * EMBED);
  k_cvt_xpw<<<(KDIR * 48 * DINNER) / 256, 256, 0, stream>>>(x_proj_w, wxp);
  k_cvt_bf16<<<128, 256, 0, stream>>>(out_proj_w, wop, EMBED * DINNER);
  k_cvt_bf16<<<256, 256, 0, stream>>>(fc1_w, wf1, FFC * EMBED);
  k_cvt_bf16<<<256, 256, 0, stream>>>(fc2_w, wf2, EMBED * FFC);

  // zero the cross-merge accumulator
  hipMemsetAsync(ym, 0, (size_t)BSZ * Lc * DINNER * sizeof(float), stream);

  // ---- OSSM branch ----
  k_gemm_ml<<<dim3(Lc / 64, (2 * DINNER) / 64, BSZ), 128, 0, stream>>>(
      wib, xb16, nullptr, xz, EMBED, (size_t)Lc * EMBED, (size_t)2 * DINNER * Lc);
  k_dwconv_act<<<(BSZ * DINNER * Lc) / 256, 256, 0, stream>>>(
      xz, conv_w, conv_b, xm, xm16, DINNER, (size_t)2 * DINNER * Lc,
      (size_t)DINNER * Lc, 0);
  k_gemm_xproj<<<dim3(Lc / 64, BSZ * KDIR), 128, 0, stream>>>(wxp, xm16, xdbl);
  k_dtproj<<<(BSZ * KDIR * DINNER * Lc) / 256, 256, 0, stream>>>(
      xdbl, dt_proj_w, dt_proj_b, delta);
  k_scan<<<BSZ * KDIR, DINNER, 0, stream>>>(xm, xdbl, delta, A_logs, Ds, ym);
  k_ln_gate<<<(BSZ * Lc) / 8, 256, 0, stream>>>(ym, xz, out_norm_g, out_norm_b, yg16);
  k_gemm_outproj<<<dim3(Lc / 64, EMBED / 64, BSZ), 128, 0, stream>>>(
      wop, yg16, x, xres);

  // ---- MixFFN branch ----
  k_ln2<<<(BSZ * Lc) / 8, 256, 0, stream>>>(xres, norm2_g, norm2_b, xn16);
  k_gemm_ml<<<dim3(Lc / 64, FFC / 64, BSZ), 128, 0, stream>>>(
      wf1, xn16, fc1_b, ffn1, EMBED, (size_t)Lc * EMBED, (size_t)FFC * Lc);
  k_dwconv_act<<<(BSZ * FFC * Lc) / 256, 256, 0, stream>>>(
      ffn1, pe_w, pe_b, nullptr, ffn2, FFC, (size_t)FFC * Lc, (size_t)FFC * Lc, 1);
  k_gemm_fc2<<<dim3(Lc / 64, EMBED / 64, BSZ), 128, 0, stream>>>(
      wf2, ffn2, xres, fc2_b, outp);
}
